// SimulatedRetriever_8555574854160
// MI455X (gfx1250) — matrix operand order repaired and checked by reading, not compiled, not run on hardware
//
#include <hip/hip_runtime.h>

// ---------------- problem constants ----------------
#define B_Q      1024
#define N_CORPUS 262144
#define QD       256
#define DM       1024
#define TOPK_K   8

// ---------------- tiling ----------------
#define QT       32                     // queries per workgroup tile
#define SLICE    512                    // keys per workgroup slice
#define NSLICES  (N_CORPUS / SLICE)     // 512
#define CAND_PER_Q (NSLICES * TOPK_K)   // 4096

typedef __attribute__((ext_vector_type(16))) _Float16 v16h;
typedef __attribute__((ext_vector_type(8)))  _Float16 h8;
typedef __attribute__((ext_vector_type(4)))  _Float16 h4;
typedef __attribute__((ext_vector_type(8)))  float    v8f;

union V16 { v16h v; struct { h8 lo; h8 hi; } p; };

// ---------------------------------------------------------------------------
// Kernel 1: L2-normalize rows of a [rows x 256] f32 matrix -> f16 (row major).
// One wave (32 lanes) per row, 8 rows per 256-thread block.
// ---------------------------------------------------------------------------
__global__ __launch_bounds__(256) void nrm_rows(const float* __restrict__ src,
                                                _Float16* __restrict__ dst) {
  const int tid  = threadIdx.x;
  const int row  = blockIdx.x * 8 + (tid >> 5);
  const int lane = tid & 31;

  const float4* p = (const float4*)(src + (size_t)row * QD);
  float4 v0 = p[lane];
  float4 v1 = p[lane + 32];
  float s = v0.x*v0.x + v0.y*v0.y + v0.z*v0.z + v0.w*v0.w
          + v1.x*v1.x + v1.y*v1.y + v1.z*v1.z + v1.w*v1.w;
  #pragma unroll
  for (int off = 16; off >= 1; off >>= 1) s += __shfl_xor(s, off, 32);

  const float inv = 1.0f / fmaxf(sqrtf(s), 1e-12f);

  _Float16* d = dst + (size_t)row * QD;
  h4 a0 = { (_Float16)(v0.x*inv), (_Float16)(v0.y*inv),
            (_Float16)(v0.z*inv), (_Float16)(v0.w*inv) };
  h4 a1 = { (_Float16)(v1.x*inv), (_Float16)(v1.y*inv),
            (_Float16)(v1.z*inv), (_Float16)(v1.w*inv) };
  *(h4*)(d + lane * 4)        = a0;
  *(h4*)(d + (lane + 32) * 4) = a1;
}

// ---------------------------------------------------------------------------
// Kernel 2: WMMA GEMM (32 queries x 512 keys, K=256 in f16) + per-slice top-8.
// grid = (NSLICES, B_Q/QT), block = 256 threads (8 waves).
// Wave w owns keys [w*64, w*64+64) of the slice: 4 key-tiles x 2 query-tiles.
// ---------------------------------------------------------------------------
__global__ __launch_bounds__(256) void sim_topk(const _Float16* __restrict__ qf16,
                                                const _Float16* __restrict__ kf16,
                                                float* __restrict__ cand_s,
                                                int*   __restrict__ cand_i) {
  __shared__ _Float16 aLds[QT][QD + 8];      // pad: row stride 264 halves (4-bank skew)
  __shared__ float    sims[QT][SLICE + 4];   // pad: row stride 516 floats

  const int tid   = threadIdx.x;
  const int lane  = tid & 31;
  const int w     = tid >> 5;
  const int slice = blockIdx.x;
  const int q0    = blockIdx.y * QT;

  // Stage normalized-query tile (32 x 256 halves) into LDS.
  for (int i = tid; i < QT * (QD / 8); i += 256) {
    const int r = i >> 5;       // row 0..31
    const int c = i & 31;       // 8-half chunk 0..31
    *(h8*)(&aLds[r][c * 8]) = *(const h8*)(qf16 + (size_t)(q0 + r) * QD + c * 8);
  }
  __syncthreads();

  const int koff    = (lane >> 4) << 3;   // A-fragment K skew: lanes 16-31 start at K+8
  const int nIdx    = lane & 15;
  const int keyBase = slice * SLICE + w * 64;

  v8f acc[4][2] = {};

  for (int kk = 0; kk < 8; ++kk) {        // K = kk*32 .. kk*32+31
    const int kb = kk * 32 + koff;
    // A fragments (16x32 f16): lane holds row (lane&15), K = {kb..kb+7, kb+16..kb+23}
    V16 a0, a1;
    a0.p.lo = *(const h8*)(&aLds[nIdx][kb]);
    a0.p.hi = *(const h8*)(&aLds[nIdx][kb + 16]);
    a1.p.lo = *(const h8*)(&aLds[nIdx + 16][kb]);
    a1.p.hi = *(const h8*)(&aLds[nIdx + 16][kb + 16]);

    #pragma unroll
    for (int t = 0; t < 4; ++t) {
      // B fragment (32x16 f16): lane holds column key=(lane&15),
      // K = {kb..kb+7, kb+16..kb+23} (same order as A).
      const size_t key = (size_t)(keyBase + t * 16 + nIdx);
      V16 b;
      b.p.lo = *(const h8*)(kf16 + key * QD + kb);
      b.p.hi = *(const h8*)(kf16 + key * QD + kb + 16);
      acc[t][0] = __builtin_amdgcn_wmma_f32_16x16x32_f16(
          false, a0.v, false, b.v, (short)0, acc[t][0], false, false);
      acc[t][1] = __builtin_amdgcn_wmma_f32_16x16x32_f16(
          false, a1.v, false, b.v, (short)0, acc[t][1], false, false);
    }
  }

  // Scatter C fragments: VGPR r, lanes 0-15 -> M=r, lanes 16-31 -> M=r+8; N=lane&15.
  const int mBase = (lane >> 4) << 3;
  #pragma unroll
  for (int t = 0; t < 4; ++t) {
    const int col = w * 64 + t * 16 + nIdx;
    #pragma unroll
    for (int r = 0; r < 8; ++r) {
      sims[mBase + r][col]      = acc[t][0][r];
      sims[mBase + r + 16][col] = acc[t][1][r];
    }
  }
  __syncthreads();

  // Per-slice top-8 extraction: 4 queries per wave, 8 argmax passes each.
  for (int j = 0; j < 4; ++j) {
    const int q = w * 4 + j;
    const size_t base = ((size_t)(q0 + q) * NSLICES + slice) * TOPK_K;
    for (int r = 0; r < TOPK_K; ++r) {
      float best = -3.0e38f; int bi = SLICE;
      #pragma unroll
      for (int p = 0; p < SLICE / 32; ++p) {
        const int i = lane + (p << 5);
        const float sv = sims[q][i];
        if (sv > best) { best = sv; bi = i; }
      }
      #pragma unroll
      for (int off = 16; off >= 1; off >>= 1) {
        const float ob = __shfl_xor(best, off, 32);
        const int   oi = __shfl_xor(bi, off, 32);
        if (ob > best || (ob == best && oi < bi)) { best = ob; bi = oi; }
      }
      if (lane == (bi & 31)) sims[q][bi] = -3.0e38f;   // retire the winner
      if (lane == 0) {
        cand_s[base + r] = best;
        cand_i[base + r] = slice * SLICE + bi;
      }
    }
  }
}

// ---------------------------------------------------------------------------
// Kernel 3: merge 4096 per-slice candidates -> final sorted top-8 per query.
// One wave per query. Scores go straight to d_out; indices to workspace.
// ---------------------------------------------------------------------------
__global__ __launch_bounds__(32) void merge_topk(const float* __restrict__ cand_s,
                                                 const int*   __restrict__ cand_i,
                                                 float* __restrict__ out_scores,
                                                 int*   __restrict__ fidx) {
  __shared__ float cs[CAND_PER_Q];
  __shared__ int   ci[CAND_PER_Q];
  const int b = blockIdx.x, lane = threadIdx.x;
  const size_t base = (size_t)b * CAND_PER_Q;
  for (int i = lane; i < CAND_PER_Q; i += 32) {
    cs[i] = cand_s[base + i];
    ci[i] = cand_i[base + i];
  }
  __syncthreads();   // single-wave workgroup -> NOP, kept for clarity

  for (int r = 0; r < TOPK_K; ++r) {
    float best = -3.0e38f; int bidx = N_CORPUS; int bslot = 0;
    for (int p = 0; p < CAND_PER_Q / 32; ++p) {
      const int i = lane + (p << 5);
      const float sv = cs[i]; const int iv = ci[i];
      if (sv > best || (sv == best && iv < bidx)) { best = sv; bidx = iv; bslot = i; }
    }
    #pragma unroll
    for (int off = 16; off >= 1; off >>= 1) {
      const float ob = __shfl_xor(best, off, 32);
      const int   oi = __shfl_xor(bidx, off, 32);
      const int   os = __shfl_xor(bslot, off, 32);
      if (ob > best || (ob == best && oi < bidx)) { best = ob; bidx = oi; bslot = os; }
    }
    if (lane == (bslot & 31)) cs[bslot] = -3.0e38f;
    if (lane == 0) {
      out_scores[b * TOPK_K + r] = best;
      fidx[b * TOPK_K + r]       = bidx;
    }
  }
}

// ---------------------------------------------------------------------------
// Kernel 4: gather corpus_values rows -> docs. One block per (query, rank),
// 256 threads x float4 = exactly one 1024-float row.
// ---------------------------------------------------------------------------
__global__ __launch_bounds__(256) void gather_docs(const float* __restrict__ values,
                                                   const int*   __restrict__ fidx,
                                                   float* __restrict__ docs) {
  const int row = blockIdx.x;                 // b*8 + rank
  const int idx = fidx[row];
  const float4* s = (const float4*)(values + (size_t)idx * DM);
  float4*       d = (float4*)(docs + (size_t)row * DM);
  d[threadIdx.x] = s[threadIdx.x];
}

// ---------------------------------------------------------------------------
extern "C" void kernel_launch(void* const* d_in, const int* in_sizes, int n_in,
                              void* d_out, int out_size, void* d_ws, size_t ws_size,
                              hipStream_t stream) {
  (void)in_sizes; (void)n_in; (void)out_size; (void)ws_size;
  const float* query  = (const float*)d_in[0];
  const float* keys   = (const float*)d_in[1];
  const float* values = (const float*)d_in[2];
  // d_in[3] = top_k scalar; output layout is fixed at K=8.

  char* ws = (char*)d_ws;
  _Float16* qf16 = (_Float16*)ws;  ws += (size_t)B_Q      * QD * sizeof(_Float16);
  _Float16* kf16 = (_Float16*)ws;  ws += (size_t)N_CORPUS * QD * sizeof(_Float16);
  float* cand_s  = (float*)ws;     ws += (size_t)B_Q * CAND_PER_Q * sizeof(float);
  int*   cand_i  = (int*)ws;       ws += (size_t)B_Q * CAND_PER_Q * sizeof(int);
  int*   fidx    = (int*)ws;

  float* docs   = (float*)d_out;                                   // [B, 8, 1024]
  float* scores = (float*)d_out + (size_t)B_Q * TOPK_K * DM;       // [B, 8]

  nrm_rows<<<B_Q / 8,      256, 0, stream>>>(query, qf16);
  nrm_rows<<<N_CORPUS / 8, 256, 0, stream>>>(keys,  kf16);

  dim3 g(NSLICES, B_Q / QT);
  sim_topk<<<g, 256, 0, stream>>>(qf16, kf16, cand_s, cand_i);

  merge_topk<<<B_Q, 32, 0, stream>>>(cand_s, cand_i, scores, fidx);
  gather_docs<<<B_Q * TOPK_K, 256, 0, stream>>>(values, fidx, docs);
}
